// MultiheadAttention_77369540870868
// MI455X (gfx1250) — matrix operand order repaired and checked by reading, not compiled, not run on hardware
//
#include <hip/hip_runtime.h>

// ---------------------------------------------------------------------------
// MI455X / gfx1250 multihead attention, bf16 WMMA (v_wmma_f32_16x16x32_bf16)
// wave32, 8-wave workgroups, 64-key flash-attention tiles, LDS-transposed V.
// All hot streaming loads forced to AS(1) => global_load_b128 (never flat).
// ---------------------------------------------------------------------------

typedef __attribute__((ext_vector_type(16))) __bf16 v16bf;
typedef __attribute__((ext_vector_type(8)))  __bf16 v8bf;
typedef __attribute__((ext_vector_type(8)))  float  v8f;

typedef __bf16 __attribute__((address_space(1))) gbf16;
typedef float  __attribute__((address_space(1))) gf32;

#define MODEL_DIM 1024
#define HEADS     16
#define HEAD_DIM  64
#define SEQ       2048
#define BATCH     2
#define MROWS     (BATCH * SEQ)   // 4096

static __device__ inline const gbf16* gptr_h(const void* base, size_t elem) {
  return (const gbf16*)(uintptr_t)((const __bf16*)base + elem);
}
static __device__ inline const gf32* gptr_f(const void* base, size_t elem) {
  return (const gf32*)(uintptr_t)((const float*)base + elem);
}
static __device__ inline v16bf gload16(const gbf16* p) {
  return *(const __attribute__((address_space(1))) v16bf*)p;
}
static __device__ inline v8bf gload8(const gbf16* p) {
  return *(const __attribute__((address_space(1))) v8bf*)p;
}
static __device__ inline v8f gloadf8(const gf32* p) {
  return *(const __attribute__((address_space(1))) v8f*)p;
}

static __device__ inline v16bf combine8(v8bf lo, v8bf hi) {
  return __builtin_shufflevector(lo, hi, 0,1,2,3,4,5,6,7,8,9,10,11,12,13,14,15);
}

static __device__ inline v8f wmma_bf16(v16bf a, v16bf b, v8f c) {
  return __builtin_amdgcn_wmma_f32_16x16x32_bf16(false, a, false, b, (short)0, c,
                                                 false, false);
}

// ---------------- f32 -> bf16 conversion (8 elements / thread) -------------
__global__ __launch_bounds__(256)
void cvt_bf16_kernel(const float* __restrict__ src, __bf16* __restrict__ dst, int n8) {
  const int i = blockIdx.x * 256 + threadIdx.x;
  if (i >= n8) return;
  v8f x = gloadf8(gptr_f(src, (size_t)i * 8));
  v8bf y;
#pragma unroll
  for (int k = 0; k < 8; ++k) y[k] = (__bf16)x[k];
  *(v8bf*)(dst + (size_t)i * 8) = y;
}

// ---------------- C[M,N] = (A[M,K] @ W[N,K]^T + bias[N]) * oscale ----------
// M=4096, N=K=1024. Block = 8 waves covering 256 rows x 64 cols.
// Wave: 32 rows (2 A-fragments) x 64 cols (4 B-fragments) -> 8 WMMA / K-step.
template<bool A_F32, bool W_F32, bool OUT_F32>
__global__ __launch_bounds__(256)
void gemm_nt_kernel(const void* __restrict__ Av, const void* __restrict__ Wp,
                    const float* __restrict__ bias, void* __restrict__ Cv,
                    float oscale) {
  const int lane = threadIdx.x & 31;
  const int wave = threadIdx.x >> 5;
  const int half = lane >> 4;
  const int l15  = lane & 15;
  const int m0 = blockIdx.y * 256 + wave * 32;
  const int n0 = blockIdx.x * 64;

  v8f acc[2][4] = {};

  const gf32*  af0 = gptr_f(Av, (size_t)(m0 + l15) * MODEL_DIM + half * 16);
  const gf32*  af1 = af0 + (size_t)16 * MODEL_DIM;
  const gbf16* ah0 = gptr_h(Av, (size_t)(m0 + l15) * MODEL_DIM + half * 16);
  const gbf16* ah1 = ah0 + (size_t)16 * MODEL_DIM;

  const gf32*  wf0 = gptr_f(Wp, (size_t)(n0 +  0 + l15) * MODEL_DIM + half * 16);
  const gf32*  wf1 = gptr_f(Wp, (size_t)(n0 + 16 + l15) * MODEL_DIM + half * 16);
  const gf32*  wf2 = gptr_f(Wp, (size_t)(n0 + 32 + l15) * MODEL_DIM + half * 16);
  const gf32*  wf3 = gptr_f(Wp, (size_t)(n0 + 48 + l15) * MODEL_DIM + half * 16);
  const gbf16* wh0 = gptr_h(Wp, (size_t)(n0 +  0 + l15) * MODEL_DIM + half * 16);
  const gbf16* wh1 = gptr_h(Wp, (size_t)(n0 + 16 + l15) * MODEL_DIM + half * 16);
  const gbf16* wh2 = gptr_h(Wp, (size_t)(n0 + 32 + l15) * MODEL_DIM + half * 16);
  const gbf16* wh3 = gptr_h(Wp, (size_t)(n0 + 48 + l15) * MODEL_DIM + half * 16);

  auto loadW = [&](const gbf16* ph, const gf32* pf) -> v16bf {
    if constexpr (W_F32) {
      v8f w0 = gloadf8(pf);
      v8f w1 = gloadf8(pf + 8);
      v16bf b;
#pragma unroll
      for (int i = 0; i < 8; ++i) { b[i] = (__bf16)w0[i]; b[8 + i] = (__bf16)w1[i]; }
      return b;
    } else {
      return combine8(gload8(ph), gload8(ph + 8));
    }
  };

  for (int kk = 0; kk < MODEL_DIM; kk += 32) {
    v16bf a0, a1;
    if constexpr (A_F32) {
      v8f lo0 = gloadf8(af0);
      v8f hi0 = gloadf8(af0 + 8);
      v8f lo1 = gloadf8(af1);
      v8f hi1 = gloadf8(af1 + 8);
      af0 += 32; af1 += 32;
#pragma unroll
      for (int i = 0; i < 8; ++i) {
        a0[i] = (__bf16)lo0[i]; a0[8 + i] = (__bf16)hi0[i];
        a1[i] = (__bf16)lo1[i]; a1[8 + i] = (__bf16)hi1[i];
      }
    } else {
      a0 = combine8(gload8(ah0), gload8(ah0 + 8));
      a1 = combine8(gload8(ah1), gload8(ah1 + 8));
      ah0 += 32; ah1 += 32;
    }
    const v16bf b0 = loadW(wh0, wf0);
    const v16bf b1 = loadW(wh1, wf1);
    const v16bf b2 = loadW(wh2, wf2);
    const v16bf b3 = loadW(wh3, wf3);
    wh0 += 32; wh1 += 32; wh2 += 32; wh3 += 32;
    wf0 += 32; wf1 += 32; wf2 += 32; wf3 += 32;

    acc[0][0] = wmma_bf16(a0, b0, acc[0][0]);
    acc[1][0] = wmma_bf16(a1, b0, acc[1][0]);
    acc[0][1] = wmma_bf16(a0, b1, acc[0][1]);
    acc[1][1] = wmma_bf16(a1, b1, acc[1][1]);
    acc[0][2] = wmma_bf16(a0, b2, acc[0][2]);
    acc[1][2] = wmma_bf16(a1, b2, acc[1][2]);
    acc[0][3] = wmma_bf16(a0, b3, acc[0][3]);
    acc[1][3] = wmma_bf16(a1, b3, acc[1][3]);
  }

#pragma unroll
  for (int t = 0; t < 2; ++t) {
#pragma unroll
    for (int g = 0; g < 4; ++g) {
      const int n = n0 + g * 16 + l15;
      const float bn = bias[n];
#pragma unroll
      for (int r = 0; r < 8; ++r) {
        const int m = m0 + t * 16 + r + 8 * half;      // C layout: M = r + 8*(lane>=16)
        const float v = (acc[t][g][r] + bn) * oscale;
        if constexpr (OUT_F32) ((float*)Cv)[(size_t)m * MODEL_DIM + n] = v;
        else ((__bf16*)Cv)[(size_t)m * MODEL_DIM + n] = (__bf16)v;
      }
    }
  }
}

// ---------------- flash attention, 64-key tiles ----------------------------
// Grid = B*H*(Sq/128) = 512 blocks of 256 threads. Wave owns 16 q-rows.
// Q is pre-scaled by 1/sqrt(HEAD_DIM) in its projection epilogue.
__global__ __launch_bounds__(256)
void attn_kernel(const __bf16* __restrict__ Q, const __bf16* __restrict__ K,
                 const __bf16* __restrict__ V, __bf16* __restrict__ O) {
  __shared__ __attribute__((aligned(64))) __bf16 VT[64][72];      // V^T tile (64k x 64d)
  __shared__ __attribute__((aligned(64))) __bf16 Pb[8][16][72];   // per-wave P tile 16x64

  const int lane = threadIdx.x & 31;
  const int wave = threadIdx.x >> 5;
  const int half = lane >> 4;
  const int l15  = lane & 15;
  const int bid  = blockIdx.x;
  const int qb = bid & 15;
  const int h  = (bid >> 4) & 15;
  const int b  = bid >> 8;
  const int rowbase = b * SEQ;

  const gbf16* qp = gptr_h(Q, (size_t)(rowbase + qb * 128 + wave * 16 + l15) * MODEL_DIM
                               + h * HEAD_DIM);
  const v16bf aq0 = combine8(gload8(qp + half * 16),      gload8(qp + 8 + half * 16));
  const v16bf aq1 = combine8(gload8(qp + 32 + half * 16), gload8(qp + 40 + half * 16));

  v8f acc[4] = {};
  float mrun[8], lrun[8];
#pragma unroll
  for (int r = 0; r < 8; ++r) { mrun[r] = -3.0e38f; lrun[r] = 0.0f; }

  // Persistent streaming pointers (advance by 64 rows per iteration).
  const size_t STEP = (size_t)64 * MODEL_DIM;
  const gbf16* kp0 = gptr_h(K, (size_t)(rowbase +  0 + l15) * MODEL_DIM + h * HEAD_DIM + half * 16);
  const gbf16* kp1 = gptr_h(K, (size_t)(rowbase + 16 + l15) * MODEL_DIM + h * HEAD_DIM + half * 16);
  const gbf16* kp2 = gptr_h(K, (size_t)(rowbase + 32 + l15) * MODEL_DIM + h * HEAD_DIM + half * 16);
  const gbf16* kp3 = gptr_h(K, (size_t)(rowbase + 48 + l15) * MODEL_DIM + h * HEAD_DIM + half * 16);

  const int vk = threadIdx.x & 63;         // V row staged by this thread
  const int vd = (threadIdx.x >> 6) * 16;  // 16-col chunk
  const gbf16* vp = gptr_h(V, (size_t)(rowbase + vk) * MODEL_DIM + h * HEAD_DIM + vd);

  for (int j = 0; j < SEQ; j += 64) {
    // ---- stage V block (64 x 64) transposed into LDS ----
    __syncthreads();
    {
      v16bf vv = gload16(vp);
#pragma unroll
      for (int i = 0; i < 16; ++i) VT[vd + i][vk] = vv[i];
    }
    __syncthreads();

    if (j + 64 < SEQ) {   // stream next K / next V blocks toward cache
      __builtin_prefetch((const void*)(uintptr_t)(kp0 + STEP), 0, 3);
      __builtin_prefetch((const void*)(uintptr_t)(vp + STEP), 0, 3);
    }

    // ---- scores: S(16x64) = Q(16x64) . K^T, 8 WMMA ----
    const v16bf bk00 = gload16(kp0);
    const v16bf bk01 = gload16(kp0 + 32);
    const v16bf bk10 = gload16(kp1);
    const v16bf bk11 = gload16(kp1 + 32);
    const v16bf bk20 = gload16(kp2);
    const v16bf bk21 = gload16(kp2 + 32);
    const v16bf bk30 = gload16(kp3);
    const v16bf bk31 = gload16(kp3 + 32);
    kp0 += STEP; kp1 += STEP; kp2 += STEP; kp3 += STEP;

    v8f s[4];
    const v8f z = {};
    s[0] = wmma_bf16(aq0, bk00, z);  s[0] = wmma_bf16(aq1, bk01, s[0]);
    s[1] = wmma_bf16(aq0, bk10, z);  s[1] = wmma_bf16(aq1, bk11, s[1]);
    s[2] = wmma_bf16(aq0, bk20, z);  s[2] = wmma_bf16(aq1, bk21, s[2]);
    s[3] = wmma_bf16(aq0, bk30, z);  s[3] = wmma_bf16(aq1, bk31, s[3]);

    // ---- online softmax over 64 keys (row reduce across 16-lane halves) ----
    float sc[8];
#pragma unroll
    for (int r = 0; r < 8; ++r) {
      const float a0 = s[0][r];
      const float a1 = s[1][r];
      const float a2 = s[2][r];
      const float a3 = s[3][r];
      float mx = fmaxf(fmaxf(a0, a1), fmaxf(a2, a3));
      mx = fmaxf(mx, __shfl_xor(mx, 1, 32));
      mx = fmaxf(mx, __shfl_xor(mx, 2, 32));
      mx = fmaxf(mx, __shfl_xor(mx, 4, 32));
      mx = fmaxf(mx, __shfl_xor(mx, 8, 32));
      const float mnew = fmaxf(mrun[r], mx);
      sc[r] = __expf(mrun[r] - mnew);
      const float p0 = __expf(a0 - mnew);
      const float p1 = __expf(a1 - mnew);
      const float p2 = __expf(a2 - mnew);
      const float p3 = __expf(a3 - mnew);
      s[0][r] = p0; s[1][r] = p1; s[2][r] = p2; s[3][r] = p3;
      float sum = (p0 + p1) + (p2 + p3);
      sum += __shfl_xor(sum, 1, 32);
      sum += __shfl_xor(sum, 2, 32);
      sum += __shfl_xor(sum, 4, 32);
      sum += __shfl_xor(sum, 8, 32);
      lrun[r] = lrun[r] * sc[r] + sum;
      mrun[r] = mnew;
    }
#pragma unroll
    for (int g = 0; g < 4; ++g)
#pragma unroll
      for (int r = 0; r < 8; ++r) acc[g][r] *= sc[r];

    // ---- P: C-layout -> A-layout via per-wave LDS tile ----
#pragma unroll
    for (int r = 0; r < 8; ++r) {
#pragma unroll
      for (int g = 0; g < 4; ++g)
        Pb[wave][r + 8 * half][16 * g + l15] = (__bf16)s[g][r];
    }
    const v16bf ap0 = combine8(*(const v8bf*)(&Pb[wave][l15][half * 16]),
                               *(const v8bf*)(&Pb[wave][l15][8 + half * 16]));
    const v16bf ap1 = combine8(*(const v8bf*)(&Pb[wave][l15][32 + half * 16]),
                               *(const v8bf*)(&Pb[wave][l15][40 + half * 16]));

    // ---- ctx += P(16x64) @ V(64x64): 8 WMMA, conflict-free LDS reads ----
#pragma unroll
    for (int g = 0; g < 4; ++g) {
      const __bf16* vrow = &VT[g * 16 + l15][0];
      const v16bf bv0 = combine8(*(const v8bf*)(vrow + half * 16),
                                 *(const v8bf*)(vrow + half * 16 + 8));
      const v16bf bv1 = combine8(*(const v8bf*)(vrow + 32 + half * 16),
                                 *(const v8bf*)(vrow + 32 + half * 16 + 8));
      acc[g] = wmma_bf16(ap0, bv0, acc[g]);
      acc[g] = wmma_bf16(ap1, bv1, acc[g]);
    }

    vp += STEP;
  }

  // ---- normalize and store ctx (bf16, merged-head layout) ----
#pragma unroll
  for (int r = 0; r < 8; ++r) {
    const float inv = 1.0f / lrun[r];
    const int m = rowbase + qb * 128 + wave * 16 + r + 8 * half;
#pragma unroll
    for (int g = 0; g < 4; ++g) {
      const int n = h * HEAD_DIM + g * 16 + l15;
      O[(size_t)m * MODEL_DIM + n] = (__bf16)(acc[g][r] * inv);
    }
  }
}

extern "C" void kernel_launch(void* const* d_in, const int* in_sizes, int n_in,
                              void* d_out, int out_size, void* d_ws, size_t ws_size,
                              hipStream_t stream) {
  (void)in_sizes; (void)n_in; (void)out_size;
  const float* query = (const float*)d_in[0];
  const float* bank  = (const float*)d_in[1];
  const float* Wq    = (const float*)d_in[2];
  const float* bq    = (const float*)d_in[3];
  const float* Wk    = (const float*)d_in[4];
  const float* bk    = (const float*)d_in[5];
  const float* Wv    = (const float*)d_in[6];
  const float* bv    = (const float*)d_in[7];
  const float* Wo    = (const float*)d_in[8];
  const float* bo    = (const float*)d_in[9];

  char* ws = (char*)d_ws;
  const size_t ACT = (size_t)MROWS * MODEL_DIM * sizeof(__bf16);       // 8 MiB
  const size_t WMT = (size_t)MODEL_DIM * MODEL_DIM * sizeof(__bf16);   // 2 MiB
  __bf16* Qb = (__bf16*)(ws);
  __bf16* Kb = (__bf16*)(ws + ACT);
  __bf16* Vb = (__bf16*)(ws + 2 * ACT);
  __bf16* Cb = (__bf16*)(ws + 3 * ACT);

  const dim3 blk(256);
  const dim3 grid(MODEL_DIM / 64, MROWS / 256);
  const dim3 agrid(BATCH * HEADS * (SEQ / 128));

  const float QSCALE = 0.125f;   // 1/sqrt(HEAD_DIM), folded into Q projection

  const size_t need_fast = 4 * ACT + 2 * ACT + 4 * WMT;  // 56 MiB
  if (ws_size >= need_fast) {
    // Fast path: one-time f32->bf16 conversion, then all-bf16 WMMA GEMMs.
    __bf16* qbf = (__bf16*)(ws + 4 * ACT);
    __bf16* bbf = (__bf16*)(ws + 5 * ACT);
    __bf16* wqb = (__bf16*)(ws + 6 * ACT);
    __bf16* wkb = (__bf16*)(ws + 6 * ACT + WMT);
    __bf16* wvb = (__bf16*)(ws + 6 * ACT + 2 * WMT);
    __bf16* wob = (__bf16*)(ws + 6 * ACT + 3 * WMT);

    const int nact8 = MROWS * MODEL_DIM / 8;             // 524288
    const int nw8   = MODEL_DIM * MODEL_DIM / 8;         // 131072
    cvt_bf16_kernel<<<dim3(nact8 / 256), blk, 0, stream>>>(query, qbf, nact8);
    cvt_bf16_kernel<<<dim3(nact8 / 256), blk, 0, stream>>>(bank,  bbf, nact8);
    cvt_bf16_kernel<<<dim3(nw8 / 256),  blk, 0, stream>>>(Wq, wqb, nw8);
    cvt_bf16_kernel<<<dim3(nw8 / 256),  blk, 0, stream>>>(Wk, wkb, nw8);
    cvt_bf16_kernel<<<dim3(nw8 / 256),  blk, 0, stream>>>(Wv, wvb, nw8);
    cvt_bf16_kernel<<<dim3(nw8 / 256),  blk, 0, stream>>>(Wo, wob, nw8);

    gemm_nt_kernel<false, false, false><<<grid, blk, 0, stream>>>(qbf, wqb, bq, Qb, QSCALE);
    gemm_nt_kernel<false, false, false><<<grid, blk, 0, stream>>>(bbf, wkb, bk, Kb, 1.0f);
    gemm_nt_kernel<false, false, false><<<grid, blk, 0, stream>>>(bbf, wvb, bv, Vb, 1.0f);
    attn_kernel<<<agrid, blk, 0, stream>>>(Qb, Kb, Vb, Cb);
    gemm_nt_kernel<false, false, true ><<<grid, blk, 0, stream>>>(Cb, wob, bo, d_out, 1.0f);
  } else {
    // Fallback: convert operands inline inside the GEMM K-loop.
    gemm_nt_kernel<true,  true, false><<<grid, blk, 0, stream>>>(query, Wq, bq, Qb, QSCALE);
    gemm_nt_kernel<true,  true, false><<<grid, blk, 0, stream>>>(bank,  Wk, bk, Kb, 1.0f);
    gemm_nt_kernel<true,  true, false><<<grid, blk, 0, stream>>>(bank,  Wv, bv, Vb, 1.0f);
    attn_kernel<<<agrid, blk, 0, stream>>>(Qb, Kb, Vb, Cb);
    gemm_nt_kernel<false, true, true ><<<grid, blk, 0, stream>>>(Cb, Wo, bo, d_out, 1.0f);
  }
}
